// MixtureOfExperts_46866683134440
// MI455X (gfx1250) — matrix operand order repaired and checked by reading, not compiled, not run on hardware
//
#include <hip/hip_runtime.h>
#include <hip/hip_bf16.h>

// ---------------------------------------------------------------------------
// MoE forward for MI455X (gfx1250, wave32, WMMA).
//   B=2, S=2048 -> T=4096 tokens, D=1024, H=4096, E=8, K=2 (top-2 routing)
// Strategy: compute only the selected experts (identical math to the dense
// reference followed by gather), bf16 WMMA for all three GEMMs, weights
// pre-converted to bf16 N-major so they stay resident in the 192MB L2.
// Round-2 change: 4-way N register blocking — one A-fragment LDS load feeds
// 4 v_wmma ops (4 accumulators/wave), cutting LDS traffic 4x.
// ---------------------------------------------------------------------------

#define T_TOK   4096      // B*S tokens
#define DIM     1024      // D
#define HID     4096      // H
#define NEXP    8         // E
#define TM      16        // token tile (WMMA M)

typedef __attribute__((ext_vector_type(8)))  float  v8f;
typedef __attribute__((ext_vector_type(16))) __bf16 v16bf;
typedef __attribute__((ext_vector_type(4)))  unsigned int u32x4;

union FragU {
    v16bf v;
    u32x4 q[2];
};

// fp32 -> bf16 (round to nearest even)
static __device__ __forceinline__ unsigned short f2bf(float f) {
    unsigned u = __float_as_uint(f);
    u += 0x7FFFu + ((u >> 16) & 1u);
    return (unsigned short)(u >> 16);
}

// A-fragment (16x32 bf16, row-major in LDS, row stride `lda` elements).
// Layout: lanes 0-15 hold row M=lane, K=k0..k0+15;
//         lanes 16-31 hold row M=lane-16, K=k0+16..k0+31.
static __device__ __forceinline__ v16bf load_a_lds(const unsigned short* A, int lda,
                                                   int k0, int lane) {
    int m  = lane & 15;
    int kb = k0 + ((lane >> 4) << 4);
    FragU f;
    f.q[0] = *(const u32x4*)(A + m * lda + kb);
    f.q[1] = *(const u32x4*)(A + m * lda + kb + 8);
    return f.v;
}

// B-fragment (32x16 bf16) from N-major (transposed) weights: Bt[n][k],
// column n contiguous in K. Lane holds column n = nbase+(lane&15); lanes 0-15
// cover K=k0..k0+15, lanes 16-31 cover K=k0+16..k0+31 (32B contiguous/lane).
static __device__ __forceinline__ v16bf load_b_nmajor(const unsigned short* Bt, size_t ldk,
                                                      int k0, int nbase, int lane) {
    size_t n  = (size_t)(nbase + (lane & 15));
    int    kk = k0 + ((lane >> 4) << 4);
    const u32x4* p = (const u32x4*)(Bt + n * ldk + kk);
    FragU f;
    f.q[0] = p[0];
    f.q[1] = p[1];
    return f.v;
}

static __device__ __forceinline__ v8f wmma_bf16(v16bf a, v16bf b, v8f c) {
    return __builtin_amdgcn_wmma_f32_16x16x32_bf16(false, a, false, b,
                                                   (short)0, c, false, false);
}

// ---------------------------------------------------------------------------
// K0: zero routing counters and the combined-output accumulator
// ---------------------------------------------------------------------------
__global__ void __launch_bounds__(256) moe_zero_kernel(float* __restrict__ combined,
                                                       int* __restrict__ counts) {
    size_t i      = (size_t)blockIdx.x * blockDim.x + threadIdx.x;
    size_t stride = (size_t)gridDim.x * blockDim.x;
    for (size_t j = i; j < (size_t)T_TOK * DIM; j += stride) combined[j] = 0.0f;
    if (i < NEXP) counts[i] = 0;
}

// ---------------------------------------------------------------------------
// K1: tiled transpose + fp32->bf16 convert.  src: [R,C] row-major fp32,
//     dst: [C,R] row-major bf16 ("N-major" weights).  batch via blockIdx.z.
// ---------------------------------------------------------------------------
__global__ void __launch_bounds__(256) cvt_transpose_kernel(const float* __restrict__ src,
                                                            unsigned short* __restrict__ dst,
                                                            int R, int C) {
    __shared__ float tile[32][33];
    const size_t bsz = (size_t)R * C;
    const float* s = src + (size_t)blockIdx.z * bsz;
    unsigned short* d = dst + (size_t)blockIdx.z * bsz;
    const int c0 = blockIdx.x * 32, r0 = blockIdx.y * 32;
    const int tx = threadIdx.x, ty = threadIdx.y;   // 32 x 8
    for (int j = 0; j < 32; j += 8) {
        int r = r0 + ty + j, c = c0 + tx;
        if (r < R && c < C) tile[ty + j][tx] = s[(size_t)r * C + c];
    }
    __syncthreads();
    for (int j = 0; j < 32; j += 8) {
        int rr = c0 + ty + j, cc = r0 + tx;      // dst is [C][R]
        if (rr < C && cc < R) d[(size_t)rr * R + cc] = f2bf(tile[tx][ty + j]);
    }
}

// ---------------------------------------------------------------------------
// K2: gating.  One wave32 per token: logits = x@Wg + bg, exact top-2 of the
// softmax (denominator cancels under renormalization), scatter token into
// per-expert lists.  Tie-break matches jax.lax.top_k (lowest index first).
// ---------------------------------------------------------------------------
__global__ void __launch_bounds__(256) moe_gate_kernel(const float* __restrict__ x,
                                                       const float* __restrict__ wg,
                                                       const float* __restrict__ bg,
                                                       int* __restrict__ counts,
                                                       int* __restrict__ lists,
                                                       float* __restrict__ wts) {
    const int tid = threadIdx.x;
    const int wave = tid >> 5, lane = tid & 31;
    const int t = blockIdx.x * 8 + wave;
    if (t >= T_TOK) return;
    const float* xr = x + (size_t)t * DIM;

    float acc[NEXP] = {};
    for (int i = 0; i < DIM / 32; ++i) {
        int d = lane + (i << 5);
        float xv = xr[d];
        const float4* wrow = (const float4*)(wg + (size_t)d * NEXP);
        float4 w0 = wrow[0], w1 = wrow[1];
        acc[0] += xv * w0.x; acc[1] += xv * w0.y;
        acc[2] += xv * w0.z; acc[3] += xv * w0.w;
        acc[4] += xv * w1.x; acc[5] += xv * w1.y;
        acc[6] += xv * w1.z; acc[7] += xv * w1.w;
    }
    for (int off = 16; off; off >>= 1)
        for (int e = 0; e < NEXP; ++e) acc[e] += __shfl_xor(acc[e], off, 32);

    if (lane == 0) {
        float l[NEXP];
        for (int e = 0; e < NEXP; ++e) l[e] = acc[e] + bg[e];
        int i0 = 0;
        for (int e = 1; e < NEXP; ++e) if (l[e] > l[i0]) i0 = e;
        int i1 = (i0 == 0) ? 1 : 0;
        for (int e = 0; e < NEXP; ++e) if (e != i0 && l[e] > l[i1]) i1 = e;
        float p1 = expf(l[i1] - l[i0]);          // p(top2)/p(top1)
        float inv = 1.0f / (1.0f + p1);
        float w0 = inv, w1 = p1 * inv;           // renormalized top-2 weights
        int p = atomicAdd(&counts[i0], 1);
        lists[i0 * T_TOK + p] = t; wts[i0 * T_TOK + p] = w0;
        p = atomicAdd(&counts[i1], 1);
        lists[i1 * T_TOK + p] = t; wts[i1 * T_TOK + p] = w1;
    }
}

// ---------------------------------------------------------------------------
// K3: fused per-expert FFN on 16-token tiles.
//   stage 1: h[16,4096] = relu(x_tile @ W1[e] + b1[e])   (h kept in 128KB LDS)
//   stage 2: combined[tok] += w * (h @ W2[e] + b2[e])     (2 commutative
//            fp32 atomics per element -> bitwise deterministic)
// 256 threads = 8 waves; wave w owns H-cols [512w,512w+512) in stage 1 and
// D-cols [128w,128w+128) in stage 2.  4 N-subtiles per k-sweep: one ds A-load
// feeds 4 v_wmma.  160KB dynamic LDS (CDNA5 WGP: 320KB) -> 2 blocks/WGP.
// ---------------------------------------------------------------------------
#define EXPERT_SMEM (TM * DIM * 2 + TM * HID * 2 + TM * 4 + TM * 4)

__global__ void __launch_bounds__(256) moe_expert_kernel(const float* __restrict__ x,
                                                         const float* __restrict__ b1,
                                                         const float* __restrict__ b2,
                                                         const unsigned short* __restrict__ w1t,
                                                         const unsigned short* __restrict__ w2t,
                                                         const int* __restrict__ counts,
                                                         const int* __restrict__ lists,
                                                         const float* __restrict__ wts,
                                                         float* __restrict__ combined) {
    extern __shared__ unsigned char smem[];
    unsigned short* xA = (unsigned short*)smem;                        // [16][1024] bf16
    unsigned short* hA = (unsigned short*)(smem + TM * DIM * 2);       // [16][4096] bf16
    int*   toks = (int*)(smem + TM * DIM * 2 + TM * HID * 2);          // [16]
    float* twt  = (float*)(toks + TM);                                 // [16]

    const int e    = blockIdx.y;
    const int cnt  = counts[e];
    const int row0 = blockIdx.x * TM;
    if (row0 >= cnt) return;

    const int tid = threadIdx.x;
    const int wave = tid >> 5, lane = tid & 31;

    if (tid < TM) {
        int idx = row0 + tid;
        toks[tid] = (idx < cnt) ? lists[e * T_TOK + idx] : -1;
        twt[tid]  = (idx < cnt) ? wts[e * T_TOK + idx] : 0.0f;
    }
    __syncthreads();

    // gather + convert the 16-token activation tile into LDS (coalesced in c)
    for (int i = tid; i < TM * DIM; i += 256) {
        int m = i >> 10, c = i & (DIM - 1);
        int tk = toks[m];
        xA[i] = (tk >= 0) ? f2bf(x[(size_t)tk * DIM + c]) : (unsigned short)0;
    }
    __syncthreads();

    const unsigned short* w1e = w1t + (size_t)e * HID * DIM;   // [H][D] bf16 (N-major)
    const unsigned short* w2e = w2t + (size_t)e * DIM * HID;   // [D][H] bf16 (N-major)

    // ---- stage 1: h = relu(x @ W1 + b1) -------------------------------
    // wave covers 512 H-cols as 8 groups of 4 16-col subtiles
    for (int g = 0; g < 8; ++g) {
        const int nb0 = wave * 512 + g * 64;
        if (nb0 + 64 < HID)   // stream-prefetch next group's weight columns
            __builtin_prefetch(w1e + (size_t)(nb0 + 64 + (lane & 15)) * DIM, 0, 1);
        v8f acc0 = {}, acc1 = {}, acc2 = {}, acc3 = {};
        for (int k0 = 0; k0 < DIM; k0 += 32) {
            v16bf a = load_a_lds(xA, DIM, k0, lane);   // shared by 4 WMMAs
            acc0 = wmma_bf16(a, load_b_nmajor(w1e, DIM, k0, nb0,      lane), acc0);
            acc1 = wmma_bf16(a, load_b_nmajor(w1e, DIM, k0, nb0 + 16, lane), acc1);
            acc2 = wmma_bf16(a, load_b_nmajor(w1e, DIM, k0, nb0 + 32, lane), acc2);
            acc3 = wmma_bf16(a, load_b_nmajor(w1e, DIM, k0, nb0 + 48, lane), acc3);
        }
        const int mh = (lane >> 4) << 3;     // rows 0..7 or 8..15
        v8f* accs[4] = { &acc0, &acc1, &acc2, &acc3 };
        for (int j = 0; j < 4; ++j) {
            const int col = nb0 + j * 16 + (lane & 15);
            const float bias = b1[e * HID + col];
            for (int r = 0; r < 8; ++r) {
                float v = (*accs[j])[r] + bias;
                v = v > 0.0f ? v : 0.0f;
                hA[(mh + r) * HID + col] = f2bf(v);
            }
        }
    }
    __syncthreads();

    // ---- stage 2: out = h @ W2 + b2 ; weighted accumulate -------------
    // wave covers 128 D-cols as 2 groups of 4 16-col subtiles
    for (int g = 0; g < 2; ++g) {
        const int nb0 = wave * 128 + g * 64;
        if (nb0 + 64 < DIM)
            __builtin_prefetch(w2e + (size_t)(nb0 + 64 + (lane & 15)) * HID, 0, 1);
        v8f acc0 = {}, acc1 = {}, acc2 = {}, acc3 = {};
        for (int k0 = 0; k0 < HID; k0 += 32) {
            v16bf a = load_a_lds(hA, HID, k0, lane);
            acc0 = wmma_bf16(a, load_b_nmajor(w2e, HID, k0, nb0,      lane), acc0);
            acc1 = wmma_bf16(a, load_b_nmajor(w2e, HID, k0, nb0 + 16, lane), acc1);
            acc2 = wmma_bf16(a, load_b_nmajor(w2e, HID, k0, nb0 + 32, lane), acc2);
            acc3 = wmma_bf16(a, load_b_nmajor(w2e, HID, k0, nb0 + 48, lane), acc3);
        }
        const int mh = (lane >> 4) << 3;
        v8f* accs[4] = { &acc0, &acc1, &acc2, &acc3 };
        for (int j = 0; j < 4; ++j) {
            const int col = nb0 + j * 16 + (lane & 15);
            const float bias = b2[e * DIM + col];
            for (int r = 0; r < 8; ++r) {
                int m = mh + r;
                int tk = toks[m];
                if (tk >= 0)
                    atomicAdd(&combined[(size_t)tk * DIM + col],
                              twt[m] * ((*accs[j])[r] + bias));
            }
        }
    }
}

// ---------------------------------------------------------------------------
// K4: final projection  out = combined @ Wp + bp   (M=4096, N=K=1024)
// ---------------------------------------------------------------------------
__global__ void __launch_bounds__(256) moe_final_kernel(const float* __restrict__ combined,
                                                        const unsigned short* __restrict__ wpt,
                                                        const float* __restrict__ bp,
                                                        float* __restrict__ out) {
    __shared__ unsigned short cA[TM * DIM];    // 32KB bf16 activation tile
    const int tid = threadIdx.x;
    const int wave = tid >> 5, lane = tid & 31;
    const int row0 = blockIdx.x * TM;

    for (int i = tid; i < TM * DIM; i += 256)
        cA[i] = f2bf(combined[(size_t)(row0 + (i >> 10)) * DIM + (i & (DIM - 1))]);
    __syncthreads();

    for (int g = 0; g < 2; ++g) {
        const int nb0 = wave * 128 + g * 64;
        v8f acc0 = {}, acc1 = {}, acc2 = {}, acc3 = {};
        for (int k0 = 0; k0 < DIM; k0 += 32) {
            v16bf a = load_a_lds(cA, DIM, k0, lane);
            acc0 = wmma_bf16(a, load_b_nmajor(wpt, DIM, k0, nb0,      lane), acc0);
            acc1 = wmma_bf16(a, load_b_nmajor(wpt, DIM, k0, nb0 + 16, lane), acc1);
            acc2 = wmma_bf16(a, load_b_nmajor(wpt, DIM, k0, nb0 + 32, lane), acc2);
            acc3 = wmma_bf16(a, load_b_nmajor(wpt, DIM, k0, nb0 + 48, lane), acc3);
        }
        const int mh = (lane >> 4) << 3;
        v8f* accs[4] = { &acc0, &acc1, &acc2, &acc3 };
        for (int j = 0; j < 4; ++j) {
            const int col = nb0 + j * 16 + (lane & 15);
            const float bias = bp[col];
            for (int r = 0; r < 8; ++r)
                out[(size_t)(row0 + mh + r) * DIM + col] = (*accs[j])[r] + bias;
        }
    }
}

// ---------------------------------------------------------------------------
extern "C" void kernel_launch(void* const* d_in, const int* in_sizes, int n_in,
                              void* d_out, int out_size, void* d_ws, size_t ws_size,
                              hipStream_t stream) {
    const float* x  = (const float*)d_in[0];
    const float* W1 = (const float*)d_in[1];
    const float* b1 = (const float*)d_in[2];
    const float* W2 = (const float*)d_in[3];
    const float* b2 = (const float*)d_in[4];
    const float* Wg = (const float*)d_in[5];
    const float* bg = (const float*)d_in[6];
    const float* Wp = (const float*)d_in[7];
    const float* bp = (const float*)d_in[8];
    float* out = (float*)d_out;

    // workspace carve-up (~147 MB total; 256B aligned slices)
    unsigned char* ws = (unsigned char*)d_ws;
    size_t off = 0;
    auto alloc = [&](size_t bytes) {
        size_t o = off;
        off += (bytes + 255) & ~(size_t)255;
        return o;
    };
    int*   counts   = (int*)(ws + alloc(NEXP * sizeof(int)));
    int*   lists    = (int*)(ws + alloc((size_t)NEXP * T_TOK * sizeof(int)));
    float* wts      = (float*)(ws + alloc((size_t)NEXP * T_TOK * sizeof(float)));
    float* combined = (float*)(ws + alloc((size_t)T_TOK * DIM * sizeof(float)));
    unsigned short* w1t = (unsigned short*)(ws + alloc((size_t)NEXP * HID * DIM * 2));
    unsigned short* w2t = (unsigned short*)(ws + alloc((size_t)NEXP * DIM * HID * 2));
    unsigned short* wpt = (unsigned short*)(ws + alloc((size_t)DIM * DIM * 2));

    // 160KB dynamic LDS (CDNA5 WGP has 320KB) — raise the per-kernel cap.
    hipFuncSetAttribute((const void*)moe_expert_kernel,
                        hipFuncAttributeMaxDynamicSharedMemorySize, EXPERT_SMEM);

    moe_zero_kernel<<<2048, 256, 0, stream>>>(combined, counts);

    // fp32 -> bf16 N-major weight conversion (one-shot HBM pass; bf16 copies
    // total ~130MB and stay resident in the 192MB L2 for the GEMMs).
    cvt_transpose_kernel<<<dim3(HID / 32, DIM / 32, NEXP), dim3(32, 8), 0, stream>>>(W1, w1t, DIM, HID);
    cvt_transpose_kernel<<<dim3(DIM / 32, HID / 32, NEXP), dim3(32, 8), 0, stream>>>(W2, w2t, HID, DIM);
    cvt_transpose_kernel<<<dim3(DIM / 32, DIM / 32, 1),    dim3(32, 8), 0, stream>>>(Wp, wpt, DIM, DIM);

    moe_gate_kernel<<<T_TOK / 8, 256, 0, stream>>>(x, Wg, bg, counts, lists, wts);

    // worst case: one expert owns all 4096 tokens -> 256 tiles; idle tiles
    // exit after a single scalar load of counts[e].
    moe_expert_kernel<<<dim3(T_TOK / TM, NEXP), 256, EXPERT_SMEM, stream>>>(
        x, b1, b2, w1t, w2t, counts, lists, wts, combined);

    moe_final_kernel<<<T_TOK / TM, 256, 0, stream>>>(combined, wpt, bp, out);
}
